// Block_3599182594843
// MI455X (gfx1250) — matrix operand order repaired and checked by reading, not compiled, not run on hardware
//
#include <hip/hip_runtime.h>

// ---------------- constants (match reference) ----------------
#define Bc 4
#define Tc 2048
#define Cc 384
#define Hc 6
#define HSc 64
#define Mrows (Bc*Tc)      // 8192
#define C4 (4*Cc)          // 1536

typedef __attribute__((ext_vector_type(16))) __bf16 v16bf;
typedef __attribute__((ext_vector_type(8)))  __bf16 v8bf;
typedef __attribute__((ext_vector_type(8)))  float  v8f;

// ---------------- cross-lane helpers (wave32) ----------------
template<int XM>
__device__ inline float swz(float x) {
    return __int_as_float(
        __builtin_amdgcn_ds_swizzle(__float_as_int(x), (XM << 10) | 0x1f));
}
__device__ inline float red16_max(float v) {
    v = fmaxf(v, swz<1>(v)); v = fmaxf(v, swz<2>(v));
    v = fmaxf(v, swz<4>(v)); v = fmaxf(v, swz<8>(v));
    return v;
}
__device__ inline float red16_sum(float v) {
    v += swz<1>(v); v += swz<2>(v); v += swz<4>(v); v += swz<8>(v);
    return v;
}
__device__ inline float red32_sum(float v) {
    v += swz<1>(v); v += swz<2>(v); v += swz<4>(v); v += swz<8>(v); v += swz<16>(v);
    return v;
}

// ---------------- WMMA fragment loaders ----------------
// A (16x32 bf16, row-major src): lane half h owns K [16h,16h+15]
__device__ inline v16bf load_a_frag(const __bf16* __restrict__ base, int lda, int lane) {
    int m  = lane & 15;
    int ch = (lane >> 4) * 16;
    const __bf16* p = base + (size_t)m * lda + ch;
    v8bf lo = *(const v8bf*)(p);
    v8bf hi = *(const v8bf*)(p + 8);
    v16bf r;
#pragma unroll
    for (int i = 0; i < 8; ++i) { r[i] = lo[i]; r[8 + i] = hi[i]; }
    return r;
}
// B (32x16 bf16) from column-major storage Bt[N][K]: lane half h owns K [16h,16h+15]
__device__ inline v16bf load_b_frag(const __bf16* __restrict__ base, int ldb, int lane) {
    int n  = lane & 15;
    int kc = (lane >> 4) * 16;
    return *(const v16bf*)(base + (size_t)n * ldb + kc);
}
__device__ inline v8f wmma_bf16(v16bf a, v16bf b, v8f c) {
    return __builtin_amdgcn_wmma_f32_16x16x32_bf16(false, a, false, b, (short)0, c,
                                                   false, false);
}

// ---------------- weight prep ----------------
// row-major (K x N) f32 -> column-major (N x K) bf16
__global__ void transpose2d_to_bf16(const float* __restrict__ in, __bf16* __restrict__ out,
                                    int K, int N) {
    int idx = blockIdx.x * 256 + threadIdx.x;
    if (idx >= K * N) return;
    int k = idx / N, n = idx % N;
    out[(size_t)n * K + k] = (__bf16)in[idx];
}
// (H, C, HS) f32 -> (H*HS, C) bf16  (col-major per combined output dim)
__global__ void transpose_qkv_to_bf16(const float* __restrict__ in, __bf16* __restrict__ out) {
    int idx = blockIdx.x * 256 + threadIdx.x;
    if (idx >= Hc * Cc * HSc) return;
    int h = idx / (Cc * HSc);
    int rem = idx % (Cc * HSc);
    int c = rem / HSc, d = rem % HSc;
    out[((size_t)h * HSc + d) * Cc + c] = (__bf16)in[idx];
}

// ---------------- LayerNorm (one wave per row, C=384 -> 12 elems/lane) ----------------
__global__ void __launch_bounds__(256)
ln_kernel(const float* __restrict__ x, const float* __restrict__ g,
          const float* __restrict__ b, __bf16* __restrict__ out, int rows) {
    int lane = threadIdx.x & 31;
    int wave = threadIdx.x >> 5;
    int row  = blockIdx.x * 8 + wave;
    if (row >= rows) return;
    const float* xr = x + (size_t)row * Cc;
    float v[12]; float s = 0.f;
#pragma unroll
    for (int j = 0; j < 12; ++j) { v[j] = xr[lane + 32 * j]; s += v[j]; }
    s = red32_sum(s);
    float mu = s * (1.0f / Cc);
    float q = 0.f;
#pragma unroll
    for (int j = 0; j < 12; ++j) { float d = v[j] - mu; q += d * d; }
    q = red32_sum(q);
    float rs = rsqrtf(q * (1.0f / Cc) + 1e-5f);
#pragma unroll
    for (int j = 0; j < 12; ++j) {
        int c = lane + 32 * j;
        out[(size_t)row * Cc + c] = (__bf16)((v[j] - mu) * rs * g[c] + b[c]);
    }
}

// ---------------- generic WMMA GEMM: C = A(MxK) * Bt(NxK)^T ----------------
// mode 0: q store  -> bf16 [b,h,t,d], scaled
// mode 1: k store  -> bf16 [b,h,t,d]
// mode 2: v store  -> bf16 [b,h,d,t]  (transposed for PV fragments)
// mode 3: f32 out[m*N+n] = acc + bias[n] + resid[m*N+n]
// mode 4: bf16 out[m*N+n] = relu(acc + bias[n])
__global__ void __launch_bounds__(256)
gemm_wmma(const __bf16* __restrict__ A, const __bf16* __restrict__ Bt,
          void* __restrict__ Out, const float* __restrict__ bias,
          const float* __restrict__ resid, int M, int N, int K, int mode, float scale) {
    int lane = threadIdx.x & 31;
    int wave = threadIdx.x >> 5;
    int m0 = (blockIdx.y * 8 + wave) * 16;
    int n0 = blockIdx.x * 64;
    if (m0 >= M) return;

    v8f acc[4] = {};
    for (int k0 = 0; k0 < K; k0 += 32) {
        v16bf a = load_a_frag(A + (size_t)m0 * K + k0, K, lane);
#pragma unroll
        for (int j = 0; j < 4; ++j) {
            v16bf bb = load_b_frag(Bt + (size_t)(n0 + 16 * j) * K + k0, K, lane);
            acc[j] = wmma_bf16(a, bb, acc[j]);
        }
    }
    int hf = lane >> 4, nn = lane & 15;
#pragma unroll
    for (int j = 0; j < 4; ++j) {
#pragma unroll
        for (int r = 0; r < 8; ++r) {
            int m = m0 + r + 8 * hf;
            int n = n0 + 16 * j + nn;
            float val = acc[j][r] * scale;
            if (mode <= 1) {
                int bq = m >> 11, t = m & (Tc - 1), hh = n >> 6, d = n & 63;
                ((__bf16*)Out)[((((size_t)bq * Hc + hh) * Tc + t) << 6) + d] = (__bf16)val;
            } else if (mode == 2) {
                int bq = m >> 11, t = m & (Tc - 1), hh = n >> 6, d = n & 63;
                ((__bf16*)Out)[(((size_t)bq * Hc + hh) * HSc + d) * Tc + t] = (__bf16)val;
            } else if (mode == 3) {
                ((float*)Out)[(size_t)m * N + n] = val + bias[n] + resid[(size_t)m * N + n];
            } else {
                float u = val + bias[n];
                ((__bf16*)Out)[(size_t)m * N + n] = (__bf16)(u > 0.f ? u : 0.f);
            }
        }
    }
}

// ---------------- causal flash attention (one wave per 16-query tile) ----------------
__global__ void __launch_bounds__(128)
attn_kernel(const __bf16* __restrict__ q, const __bf16* __restrict__ k,
            const __bf16* __restrict__ vT, __bf16* __restrict__ o) {
    __shared__ __attribute__((aligned(32))) __bf16 pbuf[4][16 * 64];
    int lane = threadIdx.x & 31;
    int wave = threadIdx.x >> 5;
    int wid  = blockIdx.x * 4 + wave;
    int bh = wid / (Tc / 16);
    int t0 = (wid % (Tc / 16)) * 16;
    int bq = bh / Hc, hh = bh % Hc;
    const __bf16* qp = q  + (size_t)bh * Tc * HSc;
    const __bf16* kp = k  + (size_t)bh * Tc * HSc;
    const __bf16* vp = vT + (size_t)bh * HSc * Tc;
    __bf16* pw = pbuf[wave];
    int hf = lane >> 4, nn = lane & 15;

    v16bf aq0 = load_a_frag(qp + (size_t)t0 * HSc + 0,  HSc, lane);
    v16bf aq1 = load_a_frag(qp + (size_t)t0 * HSc + 32, HSc, lane);

    float runmax[8], l[8];
    v8f acc[4] = {};
#pragma unroll
    for (int r = 0; r < 8; ++r) { runmax[r] = -1e30f; l[r] = 0.f; }
    int t_hi = t0 + 15;

    for (int s0 = 0; s0 <= t_hi; s0 += 64) {
        float tmax[8];
#pragma unroll
        for (int r = 0; r < 8; ++r) tmax[r] = -1e30f;
        v8f sc[4];
#pragma unroll
        for (int j = 0; j < 4; ++j) {
            int sk = s0 + 16 * j;
            v8f c;
            if (sk <= t_hi) {
                v16bf b0 = load_b_frag(kp + (size_t)sk * HSc + 0,  HSc, lane);
                v16bf b1 = load_b_frag(kp + (size_t)sk * HSc + 32, HSc, lane);
                v8f cz = {};
                c = wmma_bf16(aq0, b0, cz);
                c = wmma_bf16(aq1, b1, c);
                if (sk + 15 > t0) {   // diagonal-crossing tile: mask future keys
#pragma unroll
                    for (int r = 0; r < 8; ++r) {
                        int qi = t0 + r + 8 * hf;
                        int ki = sk + nn;
                        c[r] = (ki <= qi) ? c[r] : -1e30f;
                    }
                }
            } else {
#pragma unroll
                for (int r = 0; r < 8; ++r) c[r] = -1e30f;
            }
            sc[j] = c;
#pragma unroll
            for (int r = 0; r < 8; ++r) tmax[r] = fmaxf(tmax[r], c[r]);
        }
#pragma unroll
        for (int r = 0; r < 8; ++r) tmax[r] = red16_max(tmax[r]);

        float corr[8], tsum[8];
#pragma unroll
        for (int r = 0; r < 8; ++r) {
            float nm = fmaxf(runmax[r], tmax[r]);
            corr[r] = __expf(runmax[r] - nm);
            runmax[r] = nm;
            tsum[r] = 0.f;
        }
#pragma unroll
        for (int j = 0; j < 4; ++j) {
#pragma unroll
            for (int r = 0; r < 8; ++r) {
                float p = __expf(sc[j][r] - runmax[r]);
                tsum[r] += p;
                pw[(r + 8 * hf) * 64 + 16 * j + nn] = (__bf16)p;  // C-layout -> LDS
            }
        }
#pragma unroll
        for (int r = 0; r < 8; ++r) {
            tsum[r] = red16_sum(tsum[r]);
            l[r] = l[r] * corr[r] + tsum[r];
        }
#pragma unroll
        for (int j = 0; j < 4; ++j)
#pragma unroll
            for (int r = 0; r < 8; ++r) acc[j][r] *= corr[r];

        // reload P as A-fragments (same wave; compiler orders ds ops)
        v16bf ap0 = load_a_frag(pw + 0,  64, lane);
        v16bf ap1 = load_a_frag(pw + 32, 64, lane);
#pragma unroll
        for (int jd = 0; jd < 4; ++jd) {
            v16bf bv0 = load_b_frag(vp + (size_t)(16 * jd) * Tc + s0 + 0,  Tc, lane);
            v16bf bv1 = load_b_frag(vp + (size_t)(16 * jd) * Tc + s0 + 32, Tc, lane);
            acc[jd] = wmma_bf16(ap0, bv0, acc[jd]);
            acc[jd] = wmma_bf16(ap1, bv1, acc[jd]);
        }
    }
    // write concat-head output: o[b, t, h*64+d] bf16
#pragma unroll
    for (int jd = 0; jd < 4; ++jd) {
#pragma unroll
        for (int r = 0; r < 8; ++r) {
            int t = t0 + r + 8 * hf;
            int d = 16 * jd + nn;
            o[((size_t)bq * Tc + t) * Cc + hh * HSc + d] = (__bf16)(acc[jd][r] / l[r]);
        }
    }
}

// ---------------- host side ----------------
static inline size_t align256(size_t x) { return (x + 255) & ~(size_t)255; }

extern "C" void kernel_launch(void* const* d_in, const int* in_sizes, int n_in,
                              void* d_out, int out_size, void* d_ws, size_t ws_size,
                              hipStream_t stream) {
    (void)in_sizes; (void)n_in; (void)out_size; (void)ws_size;
    const float* x     = (const float*)d_in[0];
    const float* ln1_g = (const float*)d_in[1];
    const float* ln1_b = (const float*)d_in[2];
    const float* Wq    = (const float*)d_in[3];
    const float* Wk    = (const float*)d_in[4];
    const float* Wv    = (const float*)d_in[5];
    const float* Wp    = (const float*)d_in[6];
    const float* bp    = (const float*)d_in[7];
    const float* ln2_g = (const float*)d_in[8];
    const float* ln2_b = (const float*)d_in[9];
    const float* W1    = (const float*)d_in[10];
    const float* b1    = (const float*)d_in[11];
    const float* W2    = (const float*)d_in[12];
    const float* b2    = (const float*)d_in[13];
    float* out = (float*)d_out;

    char* ws = (char*)d_ws;
    size_t off = 0;
    auto alloc = [&](size_t bytes) -> char* {
        char* p = ws + off;
        off += align256(bytes);
        return p;
    };
    const size_t MC = (size_t)Mrows * Cc;          // 3,145,728
    __bf16* h1  = (__bf16*)alloc(MC * 2);
    __bf16* qb  = (__bf16*)alloc(MC * 2);
    __bf16* kb  = (__bf16*)alloc(MC * 2);
    __bf16* vtb = (__bf16*)alloc(MC * 2);
    __bf16* ob  = (__bf16*)alloc(MC * 2);
    float*  x2  = (float*) alloc(MC * 4);
    __bf16* h2  = (__bf16*)alloc(MC * 2);
    __bf16* act = (__bf16*)alloc((size_t)Mrows * C4 * 2);
    __bf16* wqT = (__bf16*)alloc((size_t)Cc * Cc * 2);
    __bf16* wkT = (__bf16*)alloc((size_t)Cc * Cc * 2);
    __bf16* wvT = (__bf16*)alloc((size_t)Cc * Cc * 2);
    __bf16* wpT = (__bf16*)alloc((size_t)Cc * Cc * 2);
    __bf16* w1T = (__bf16*)alloc((size_t)Cc * C4 * 2);
    __bf16* w2T = (__bf16*)alloc((size_t)C4 * Cc * 2);

    // weight prep
    int nqkv = Hc * Cc * HSc;
    transpose_qkv_to_bf16<<<(nqkv + 255) / 256, 256, 0, stream>>>(Wq, wqT);
    transpose_qkv_to_bf16<<<(nqkv + 255) / 256, 256, 0, stream>>>(Wk, wkT);
    transpose_qkv_to_bf16<<<(nqkv + 255) / 256, 256, 0, stream>>>(Wv, wvT);
    transpose2d_to_bf16<<<(Cc * Cc + 255) / 256, 256, 0, stream>>>(Wp, wpT, Cc, Cc);
    transpose2d_to_bf16<<<(Cc * C4 + 255) / 256, 256, 0, stream>>>(W1, w1T, Cc, C4);
    transpose2d_to_bf16<<<(C4 * Cc + 255) / 256, 256, 0, stream>>>(W2, w2T, C4, Cc);

    // LN1
    ln_kernel<<<Mrows / 8, 256, 0, stream>>>(x, ln1_g, ln1_b, h1, Mrows);

    // QKV projections (q pre-scaled by HS^-0.5 = 0.125)
    dim3 gC(Cc / 64, Mrows / 128);
    gemm_wmma<<<gC, 256, 0, stream>>>(h1, wqT, qb,  nullptr, nullptr, Mrows, Cc, Cc, 0, 0.125f);
    gemm_wmma<<<gC, 256, 0, stream>>>(h1, wkT, kb,  nullptr, nullptr, Mrows, Cc, Cc, 1, 1.0f);
    gemm_wmma<<<gC, 256, 0, stream>>>(h1, wvT, vtb, nullptr, nullptr, Mrows, Cc, Cc, 2, 1.0f);

    // causal flash attention: B*H*(T/16) = 3072 waves, 4 per block
    attn_kernel<<<(Bc * Hc * (Tc / 16)) / 4, 128, 0, stream>>>(qb, kb, vtb, ob);

    // output projection + residual
    gemm_wmma<<<gC, 256, 0, stream>>>(ob, wpT, x2, bp, x, Mrows, Cc, Cc, 3, 1.0f);

    // LN2
    ln_kernel<<<Mrows / 8, 256, 0, stream>>>(x2, ln2_g, ln2_b, h2, Mrows);

    // FFN
    dim3 g1(C4 / 64, Mrows / 128);
    gemm_wmma<<<g1, 256, 0, stream>>>(h2, w1T, act, b1, nullptr, Mrows, C4, Cc, 4, 1.0f);
    gemm_wmma<<<gC, 256, 0, stream>>>(act, w2T, out, b2, x2, Mrows, Cc, C4, 3, 1.0f);
}
